// DeepseekV3Attention_33638183862741
// MI455X (gfx1250) — matrix operand order repaired and checked by reading, not compiled, not run on hardware
//
#include <hip/hip_runtime.h>
#include <math.h>

#define T_SEQ 2048
#define HID   4096
#define NH    32
#define QLORA 1536
#define KVL   512
#define DNOPE 128
#define DROPE 64
#define DQK   192
#define DV    128
#define DEFF  576   // KVL + DROPE

typedef __attribute__((ext_vector_type(16))) __bf16          v16bf;
typedef __attribute__((ext_vector_type(16))) unsigned short  v16u;
typedef __attribute__((ext_vector_type(8)))  float           v8f;

struct U2x16B { uint4 lo, hi; };   // 32 bytes = one v16bf fragment

__device__ __forceinline__ unsigned short f2bf(float f) {
    unsigned u = __builtin_bit_cast(unsigned, f);
    u += 0x7FFFu + ((u >> 16) & 1u);           // round-to-nearest-even
    return (unsigned short)(u >> 16);
}
__device__ __forceinline__ float bf2f(unsigned short h) {
    return __builtin_bit_cast(float, (unsigned)h << 16);
}
__device__ __forceinline__ v16bf frag16(const unsigned short* p0, const unsigned short* p1) {
    U2x16B u{ *reinterpret_cast<const uint4*>(p0), *reinterpret_cast<const uint4*>(p1) };
    return __builtin_bit_cast(v16bf, u);
}

// ---------------------------------------------------------------------------
// elementwise f32 -> bf16
// ---------------------------------------------------------------------------
__global__ __launch_bounds__(256)
void cvt_bf16(const float* __restrict__ in, unsigned short* __restrict__ out, long n)
{
    long i = (long)blockIdx.x * 256 + threadIdx.x;
    if (i < n) out[i] = f2bf(in[i]);
}

// w_uk [r][h][d] f32 -> uk_t [h][d][r] bf16
__global__ void transpose_uk(const float* __restrict__ w_uk, unsigned short* __restrict__ uk_t)
{
    long i = (long)blockIdx.x * blockDim.x + threadIdx.x;
    if (i >= (long)KVL * NH * DNOPE) return;
    long r = i >> 12;                 // / (NH*DNOPE)
    long rem = i & 4095;
    long h = rem >> 7, d = rem & 127;
    uk_t[(h * DNOPE + d) * KVL + r] = f2bf(w_uk[i]);
}

// ---------------------------------------------------------------------------
// Batched GEMM: C[M,N] = alpha * A[M,K] @ B[K,N]; A,B bf16, f32 accumulate,
// f32 or bf16 out.  64x64 block tile, 4 waves, Kstep = 32.
// A tile stored [m][k]; B tile stored transposed [n][k] so every WMMA fragment
// is two aligned 16B LDS loads.
// ---------------------------------------------------------------------------
template<bool CBF>
__global__ __launch_bounds__(128)
void gemm_wmma(const unsigned short* __restrict__ A, const unsigned short* __restrict__ B,
               void* __restrict__ Cv, int N, int K,
               long lda, long ldb, long ldc,
               long sA, long sB, long sC, float alpha)
{
    __shared__ __align__(16) unsigned short aT[64][32];   // [m][k]
    __shared__ __align__(16) unsigned short bT[64][40];   // [n][k], 80B pitch
    const int tid  = threadIdx.x;
    const int w    = tid >> 5;
    const int lane = tid & 31;
    const int l15  = lane & 15, hi = lane >> 4;
    const long m0 = (long)blockIdx.y * 64;
    const long n0 = (long)blockIdx.x * 64;
    A += (long)blockIdx.z * sA;
    B += (long)blockIdx.z * sB;

    v8f acc[4];
#pragma unroll
    for (int i = 0; i < 4; ++i) acc[i] = {};

    for (int kk = 0; kk < K; kk += 32) {
        __syncthreads();
#pragma unroll
        for (int i = 0; i < 16; ++i) {           // A tile 64x32, contiguous per thread
            int idx = tid * 16 + i;
            int m = idx >> 5, k = idx & 31;
            aT[m][k] = A[(m0 + m) * lda + kk + k];
        }
#pragma unroll
        for (int i = 0; i < 16; ++i) {           // B tile 32x64 -> transposed store
            int idx = tid * 16 + i;
            int k = idx >> 6, n = idx & 63;
            bT[n][k] = B[(long)(kk + k) * ldb + n0 + n];
        }
        if (kk + 32 < K) {                       // prefetch next K tile
            __builtin_prefetch(&A[(m0 + ((tid * 16) >> 5)) * lda + kk + 32], 0, 0);
            __builtin_prefetch(&B[(long)(kk + 32 + (tid >> 3)) * ldb + n0], 0, 0);
        }
        __syncthreads();

        const unsigned short* ar = &aT[16 * w + l15][0];
        v16bf aF = frag16(ar + hi * 8, ar + 16 + hi * 8);
#pragma unroll
        for (int nt = 0; nt < 4; ++nt) {
            const unsigned short* br = &bT[nt * 16 + l15][hi * 8];
            v16bf bF = frag16(br, br + 16);
            acc[nt] = __builtin_amdgcn_wmma_f32_16x16x32_bf16(
                false, aF, false, bF, (short)0, acc[nt], false, false);
        }
    }

#pragma unroll
    for (int nt = 0; nt < 4; ++nt)
#pragma unroll
        for (int v = 0; v < 8; ++v) {
            long r = m0 + 16 * w + v + 8 * hi;
            long c = n0 + nt * 16 + l15;
            float val = alpha * acc[nt][v];
            if constexpr (CBF)
                ((unsigned short*)Cv)[(long)blockIdx.z * sC + r * ldc + c] = f2bf(val);
            else
                ((float*)Cv)[(long)blockIdx.z * sC + r * ldc + c] = val;
        }
}

// ---------------------------------------------------------------------------
template<bool OBF>
__global__ __launch_bounds__(256)
void rmsnorm_rows(const float* __restrict__ in, const float* __restrict__ scale,
                  void* __restrict__ out, int C)
{
    int t = blockIdx.x, tid = threadIdx.x;
    const float* row = in + (long)t * C;
    __shared__ float red[256];
    float ss = 0.f;
    for (int c = tid; c < C; c += 256) { float v = row[c]; ss += v * v; }
    red[tid] = ss; __syncthreads();
    for (int s = 128; s > 0; s >>= 1) { if (tid < s) red[tid] += red[tid + s]; __syncthreads(); }
    float inv = rsqrtf(red[0] / (float)C + 1e-6f);
    for (int c = tid; c < C; c += 256) {
        float v = row[c] * inv * scale[c];
        if constexpr (OBF) ((unsigned short*)out)[(long)t * C + c] = f2bf(v);
        else               ((float*)out)[(long)t * C + c] = v;
    }
}

// kv_a row (f32) -> Keff bf16 row: rmsnorm(first 512) | rope(last 64)
__global__ __launch_bounds__(128)
void pack_keff(const float* __restrict__ kv_a, const float* __restrict__ kscale,
               const int* __restrict__ pos, unsigned short* __restrict__ Keff)
{
    int t = blockIdx.x, tid = threadIdx.x;
    const float* row = kv_a + (long)t * DEFF;
    __shared__ float red[128];
    float ss = 0.f;
    for (int c = tid; c < KVL; c += 128) { float v = row[c]; ss += v * v; }
    red[tid] = ss; __syncthreads();
    for (int s = 64; s > 0; s >>= 1) { if (tid < s) red[tid] += red[tid + s]; __syncthreads(); }
    float inv = rsqrtf(red[0] / (float)KVL + 1e-6f);
    unsigned short* out = Keff + (long)t * DEFF;
    for (int c = tid; c < KVL; c += 128) out[c] = f2bf(row[c] * inv * kscale[c]);
    if (tid < 32) {
        int j = tid;
        float invf = __powf(10000.0f, -(float)j / 32.0f);
        float ang = (float)pos[t] * invf;
        float sn, cs; __sincosf(ang, &sn, &cs);
        float x1 = row[KVL + 2 * j], x2 = row[KVL + 2 * j + 1];
        out[KVL + 2 * j]     = f2bf(x1 * cs - x2 * sn);
        out[KVL + 2 * j + 1] = f2bf(x1 * sn + x2 * cs);
    }
}

// q rope slice (bf16) -> Qeff[h][t][512:576] bf16, pre-scaled by 1/sqrt(192)
__global__ __launch_bounds__(256)
void pack_qrope(const unsigned short* __restrict__ q, const int* __restrict__ pos,
                unsigned short* __restrict__ Qeff, float scale)
{
    int t = blockIdx.x;
    float p = (float)pos[t];
    for (int i = threadIdx.x; i < NH * 32; i += 256) {
        int h = i >> 5, j = i & 31;
        float invf = __powf(10000.0f, -(float)j / 32.0f);
        float ang = p * invf;
        float sn, cs; __sincosf(ang, &sn, &cs);
        const unsigned short* src = q + ((long)t * NH + h) * DQK + DNOPE;
        float x1 = bf2f(src[2 * j]), x2 = bf2f(src[2 * j + 1]);
        unsigned short* dst = Qeff + ((long)h * T_SEQ + t) * DEFF + KVL;
        dst[2 * j]     = f2bf((x1 * cs - x2 * sn) * scale);
        dst[2 * j + 1] = f2bf((x1 * sn + x2 * cs) * scale);
    }
}

// ---------------------------------------------------------------------------
// Flash attention: block = 4 waves x 16 query rows, stream 32-key tiles.
// Qeff[h][t][576] bf16 (pre-scaled), Keff[t][576] bf16 (V = cols 0..511).
// o_lat[t][h][512] bf16.
// ---------------------------------------------------------------------------
__global__ __launch_bounds__(128)
void flash_attn(const unsigned short* __restrict__ Qeff,
                const unsigned short* __restrict__ Keff,
                unsigned short* __restrict__ o_lat)
{
    __shared__ __align__(16) unsigned short vT[KVL][40];     // V^T tile [vdim][key], 80B pitch
    __shared__ __align__(16) unsigned short pBuf[4][16][40]; // per-wave P transpose buffer
    const int h  = blockIdx.y;
    const int q0 = blockIdx.x * 64;
    const int w  = threadIdx.x >> 5;
    const int lane = threadIdx.x & 31;
    const int l15 = lane & 15, hi = lane >> 4;
    const int rowBase = q0 + 16 * w;

    v8f O[32];
#pragma unroll
    for (int i = 0; i < 32; ++i) O[i] = {};
    float mrow[8], lrow[8];
#pragma unroll
    for (int v = 0; v < 8; ++v) { mrow[v] = -1e30f; lrow[v] = 0.f; }

    const unsigned short* Qr = Qeff + ((size_t)h * T_SEQ + (rowBase + l15)) * DEFF;

    for (int kb = 0; kb < q0 + 64; kb += 32) {
        __syncthreads();
        for (int i = threadIdx.x; i < 32 * KVL; i += 128) {   // stage V^T
            int kk = i >> 9, n = i & (KVL - 1);
            vT[n][kk] = Keff[(size_t)(kb + kk) * DEFF + n];
        }
        __syncthreads();

        // S = Qeff(16x576) @ Keff_tile^T(576x32), two 16-key subtiles
        v8f S0 = {}, S1 = {};
        const unsigned short* K0 = Keff + (size_t)(kb + l15) * DEFF;
        const unsigned short* K1 = K0 + (size_t)16 * DEFF;
#pragma unroll
        for (int kc = 0; kc < DEFF / 32; ++kc) {
            v16bf aF = frag16(Qr + kc * 32 + hi * 8, Qr + kc * 32 + 16 + hi * 8);
            v16bf b0 = frag16(K0 + kc * 32 + hi * 8, K0 + kc * 32 + 16 + hi * 8);
            v16bf b1 = frag16(K1 + kc * 32 + hi * 8, K1 + kc * 32 + 16 + hi * 8);
            S0 = __builtin_amdgcn_wmma_f32_16x16x32_bf16(false, aF, false, b0, (short)0, S0, false, false);
            S1 = __builtin_amdgcn_wmma_f32_16x16x32_bf16(false, aF, false, b1, (short)0, S1, false, false);
        }

        // online softmax; row m = v + 8*hi, key col = l15 (+16 for subtile 1)
        float corr[8];
#pragma unroll
        for (int v = 0; v < 8; ++v) {
            int r = rowBase + v + 8 * hi;
            float s0 = (kb + l15 <= r)      ? S0[v] : -1e30f;
            float s1 = (kb + 16 + l15 <= r) ? S1[v] : -1e30f;
            float mx = fmaxf(s0, s1);
#pragma unroll
            for (int o = 1; o < 16; o <<= 1) mx = fmaxf(mx, __shfl_xor(mx, o, 32));
            float mnew = fmaxf(mrow[v], mx);
            float p0 = __expf(s0 - mnew), p1 = __expf(s1 - mnew);
            float c  = __expf(mrow[v] - mnew);
            float ps = p0 + p1;
#pragma unroll
            for (int o = 1; o < 16; o <<= 1) ps += __shfl_xor(ps, o, 32);
            lrow[v] = lrow[v] * c + ps;
            mrow[v] = mnew;
            corr[v] = c;
            pBuf[w][v + 8 * hi][l15]      = f2bf(p0);
            pBuf[w][v + 8 * hi][16 + l15] = f2bf(p1);
        }
#pragma unroll
        for (int vt = 0; vt < 32; ++vt)
#pragma unroll
            for (int v = 0; v < 8; ++v) O[vt][v] *= corr[v];

        // P as A-fragment (wave-internal LDS round trip)
        const unsigned short* pr = &pBuf[w][l15][0];
        v16bf pF = frag16(pr + hi * 8, pr + 16 + hi * 8);
#pragma unroll
        for (int vt = 0; vt < 32; ++vt) {        // O += P(16x32) @ V(32x512)
            const unsigned short* vr = &vT[vt * 16 + l15][hi * 8];
            v16bf bF = frag16(vr, vr + 16);
            O[vt] = __builtin_amdgcn_wmma_f32_16x16x32_bf16(false, pF, false, bF, (short)0, O[vt], false, false);
        }
    }

#pragma unroll
    for (int v = 0; v < 8; ++v) {
        int r = rowBase + v + 8 * hi;
        float inv = 1.0f / lrow[v];
        unsigned short* dst = o_lat + ((size_t)r * NH + h) * KVL;
#pragma unroll
        for (int vt = 0; vt < 32; ++vt) dst[vt * 16 + l15] = f2bf(O[vt][v] * inv);
    }
}

// ---------------------------------------------------------------------------
extern "C" void kernel_launch(void* const* d_in, const int* in_sizes, int n_in,
                              void* d_out, int out_size, void* d_ws, size_t ws_size,
                              hipStream_t stream)
{
    (void)in_sizes; (void)n_in; (void)out_size; (void)ws_size;
    const float* hidden = (const float*)d_in[0];
    const int*   pos    = (const int*)d_in[1];
    const float* w_qa   = (const float*)d_in[2];
    const float* qa_ln  = (const float*)d_in[3];
    const float* w_qb   = (const float*)d_in[4];
    const float* w_kva  = (const float*)d_in[5];
    const float* kva_ln = (const float*)d_in[6];
    const float* w_uk   = (const float*)d_in[7];
    const float* w_uv   = (const float*)d_in[8];
    const float* w_o    = (const float*)d_in[9];
    float* out = (float*)d_out;

    char* ws = (char*)d_ws;
    size_t off = 0;
    auto take = [&](size_t bytes) -> char* {
        char* p = ws + off;
        off += (bytes + 255) & ~(size_t)255;
        return p;
    };
    typedef unsigned short u16;

    // persistent
    u16* Keff    = (u16*)take((size_t)T_SEQ * DEFF * 2);
    u16* Qeff    = (u16*)take((size_t)NH * T_SEQ * DEFF * 2);
    u16* w_qa_b  = (u16*)take((size_t)HID * QLORA * 2);
    u16* w_qb_b  = (u16*)take((size_t)QLORA * NH * DQK * 2);
    u16* w_kva_b = (u16*)take((size_t)HID * DEFF * 2);
    u16* uk_t_b  = (u16*)take((size_t)NH * DNOPE * KVL * 2);
    u16* w_uv_b  = (u16*)take((size_t)KVL * NH * DV * 2);
    u16* w_o_b   = (u16*)take((size_t)NH * DV * HID * 2);
    size_t overlay = off;
    // phase-1 overlay (all dead before flash writes o_lat)
    u16*   hid_b = (u16*)take((size_t)T_SEQ * HID * 2);
    float* q_a   = (float*)take((size_t)T_SEQ * QLORA * 4);
    u16*   q_c_b = (u16*)take((size_t)T_SEQ * QLORA * 2);
    u16*   qbuf  = (u16*)take((size_t)T_SEQ * NH * DQK * 2);
    float* kv_a  = (float*)take((size_t)T_SEQ * DEFF * 4);
    // phase-2 overlay
    off = overlay;
    u16* o_lat = (u16*)take((size_t)T_SEQ * NH * KVL * 2);
    u16* o_v   = (u16*)take((size_t)T_SEQ * NH * DV * 2);

    const float scale = 1.0f / sqrtf((float)DQK);
    dim3 blk(128);
    auto cvt = [&](const float* src, u16* dst, long n) {
        cvt_bf16<<<dim3((unsigned)((n + 255) / 256)), dim3(256), 0, stream>>>(src, dst, n);
    };

    // 0. one-time bf16 conversions (bandwidth-trivial; L2 resident afterwards)
    cvt(hidden, hid_b,  (long)T_SEQ * HID);
    cvt(w_qa,   w_qa_b, (long)HID * QLORA);
    cvt(w_qb,   w_qb_b, (long)QLORA * NH * DQK);
    cvt(w_kva,  w_kva_b,(long)HID * DEFF);
    cvt(w_uv,   w_uv_b, (long)KVL * NH * DV);
    cvt(w_o,    w_o_b,  (long)NH * DV * HID);
    // 1. transpose w_uk -> uk_t [h][d][r] (bf16)
    {
        long n = (long)KVL * NH * DNOPE;
        transpose_uk<<<dim3((unsigned)((n + 255) / 256)), dim3(256), 0, stream>>>(w_uk, uk_t_b);
    }
    // 2. q_a = hidden @ w_qa      [2048,1536] K=4096 (f32 out)
    gemm_wmma<false><<<dim3(QLORA / 64, T_SEQ / 64, 1), blk, 0, stream>>>(
        hid_b, w_qa_b, q_a, QLORA, HID, HID, QLORA, QLORA, 0, 0, 0, 1.0f);
    // 3. q_c = rmsnorm(q_a)  (bf16 out)
    rmsnorm_rows<true><<<T_SEQ, 256, 0, stream>>>(q_a, qa_ln, q_c_b, QLORA);
    // 4. q = q_c @ w_qb           [2048,6144] K=1536 (bf16 out)
    gemm_wmma<true><<<dim3(NH * DQK / 64, T_SEQ / 64, 1), blk, 0, stream>>>(
        q_c_b, w_qb_b, qbuf, NH * DQK, QLORA, QLORA, NH * DQK, NH * DQK, 0, 0, 0, 1.0f);
    // 5. kv_a = hidden @ w_kva    [2048,576] K=4096 (f32 out, rmsnorm needs precision)
    gemm_wmma<false><<<dim3(DEFF / 64, T_SEQ / 64, 1), blk, 0, stream>>>(
        hid_b, w_kva_b, kv_a, DEFF, HID, HID, DEFF, DEFF, 0, 0, 0, 1.0f);
    // 6. Keff = [rmsnorm(kv_a[:,:512]) | rope(kv_a[:,512:])]  (bf16)
    pack_keff<<<T_SEQ, 128, 0, stream>>>(kv_a, kva_ln, pos, Keff);
    // 7. Qeff[:, :512] = scale * q_nope @ w_uk^T   (batched per head, bf16 out)
    gemm_wmma<true><<<dim3(KVL / 64, T_SEQ / 64, NH), blk, 0, stream>>>(
        qbuf, uk_t_b, Qeff, KVL, DNOPE,
        (long)NH * DQK, KVL, DEFF,
        (long)DQK, (long)DNOPE * KVL, (long)T_SEQ * DEFF, scale);
    // 8. Qeff[:, 512:576] = scale * rope(q_rope)
    pack_qrope<<<T_SEQ, 256, 0, stream>>>(qbuf, pos, Qeff, scale);
    // 9. flash attention -> o_lat [t][h][512] bf16
    flash_attn<<<dim3(T_SEQ / 64, NH), blk, 0, stream>>>(Qeff, Keff, o_lat);
    // 10. o_v = o_lat @ w_uv      (batched per head, bf16 out)
    gemm_wmma<true><<<dim3(DV / 64, T_SEQ / 64, NH), blk, 0, stream>>>(
        o_lat, w_uv_b, o_v, DV, KVL,
        (long)NH * KVL, (long)NH * DV, (long)NH * DV,
        (long)KVL, (long)DV, (long)DV, 1.0f);
    // 11. out = o_v @ w_o         [2048,4096] K=4096 (f32 out to d_out)
    gemm_wmma<false><<<dim3(HID / 64, T_SEQ / 64, 1), blk, 0, stream>>>(
        o_v, w_o_b, out, HID, NH * DV, NH * DV, HID, HID, 0, 0, 0, 1.0f);
}
